// DownstreamSingleAttentionModel_50173807952800
// MI455X (gfx1250) — matrix operand order repaired and checked by reading, not compiled, not run on hardware
//
#include <hip/hip_runtime.h>
#include <hip/hip_bf16.h>
#include <math.h>

typedef __attribute__((ext_vector_type(16))) _Float16 v16h;
typedef __attribute__((ext_vector_type(8)))  _Float16 v8h;
typedef __attribute__((ext_vector_type(8)))  float    v8f;

#define NB    2
#define LSEQ  1024
#define CDIM  256
#define HH    64
#define EE    16
#define CLW   64
#define LMID  960                 // L - CL
#define MROWS (NB*LSEQ)           // 2048
#define QK    (HH*EE)             // 1024
#define OUTC  (2*QK + HH)         // 2112 fused output columns (q | k | v)
#define BPAD  264                 // LDS row stride in halves (16B aligned, bank-spread)

#ifndef USE_ASYNC_LDS
#define USE_ASYNC_LDS 1
#endif

// ---------------------------------------------------------------------------
// Kernel 0: one-shot f32 -> f16 conversion of X and fused [Wq;Wk;Wv].
// ---------------------------------------------------------------------------
__global__ __launch_bounds__(256)
void to_half(const float* __restrict__ x,
             const float* __restrict__ Wq, const float* __restrict__ Wk,
             const float* __restrict__ Wv,
             _Float16* __restrict__ xh, _Float16* __restrict__ wh)
{
    const int total = MROWS*CDIM + OUTC*CDIM;
    for (int i = blockIdx.x*blockDim.x + threadIdx.x; i < total;
         i += gridDim.x*blockDim.x) {
        if (i < MROWS*CDIM) {
            xh[i] = (_Float16)x[i];
        } else {
            const int j = i - MROWS*CDIM;
            const int c = j / CDIM, k = j - c*CDIM;
            float v;
            if (c < QK)        v = Wq[c*CDIM + k];
            else if (c < 2*QK) v = Wk[(c - QK)*CDIM + k];
            else               v = Wv[(c - 2*QK)*CDIM + k];
            wh[j] = (_Float16)v;
        }
    }
}

// ---------------------------------------------------------------------------
// Kernel 1: fused QKV projection.  Block = 8 waves sharing one 64-col group;
// B panel (64x256 f16, 32KB) staged into LDS once per block via async copy.
// Each wave: one 16-row tile, 4 accumulators, 8 k-steps of 32 (f16 WMMA).
// B fragments use 4 distinct register sets so all 8 ds_load_b128 issue in one
// clause and WMMA overlaps LDS latency via partial dscnt waits.
// ---------------------------------------------------------------------------
__global__ __launch_bounds__(256)
void qkv_gemm(const _Float16* __restrict__ xh, const _Float16* __restrict__ wh,
              const float* __restrict__ bq, const float* __restrict__ bk,
              const float* __restrict__ bv,
              float* __restrict__ Q, float* __restrict__ Kb, float* __restrict__ V)
{
    __shared__ _Float16 ldsB[64 * BPAD];          // padded 64 x 256 panel
    const int colGrp   = blockIdx.x % (OUTC/64);  // 0..32
    const int rowBlock = blockIdx.x / (OUTC/64);  // 0..15
    const int c0 = colGrp * 64;

    // ---- stage B panel: 2048 chunks of 16B, 8 per thread ----
    const _Float16* wsrc = wh + (size_t)c0 * CDIM;
    #pragma unroll
    for (int i = 0; i < 8; ++i) {
        const int q   = i * 256 + threadIdx.x;
        const int row = q >> 5;                   // 0..63
        const int wi  = (q & 31) * 8;             // half offset within row
        const _Float16* gp = wsrc + row * CDIM + wi;
        _Float16*       lp = ldsB + row * BPAD + wi;
#if USE_ASYNC_LDS
        asm volatile("global_load_async_to_lds_b128 %0, %1, off"
                     :: "v"((unsigned)(size_t)lp),
                        "v"((unsigned long long)(size_t)gp)
                     : "memory");
#else
        *(uint4*)lp = *(const uint4*)gp;
#endif
    }
#if USE_ASYNC_LDS
    asm volatile("s_wait_asynccnt 0x0" ::: "memory");
#endif
    __syncthreads();

    const int lane = threadIdx.x & 31;
    const int wv   = threadIdx.x >> 5;
    const int m0   = (rowBlock * 8 + wv) * 16;    // 16-row tile per wave
    const int nsel = lane & 15;
    const int aoff = (lane & 16) ? 8  : 0;        // A: K sub-off / D: row sub-off
    const int boff = (lane & 16) ? 8  : 0;        // B: K sub-off

    const _Float16* xrow = xh  + (size_t)(m0 + nsel) * CDIM;
    const _Float16* brow = ldsB + nsel * BPAD;

    v8f acc[4] = {};
    for (int kk = 0; kk < CDIM; kk += 32) {
        const v8h alo = *(const v8h*)(xrow + kk + aoff);
        const v8h ahi = *(const v8h*)(xrow + kk + aoff + 16);
        const v16h a = __builtin_shufflevector(alo, ahi,
                        0,1,2,3,4,5,6,7,8,9,10,11,12,13,14,15);

        // load all four B fragments first (distinct regs -> one ds clause)
        v16h bfrag[4];
        #pragma unroll
        for (int ct = 0; ct < 4; ++ct) {
            const _Float16* bp = brow + ct * (16*BPAD) + kk + boff;
            const v8h blo = *(const v8h*)bp;
            const v8h bhi = *(const v8h*)(bp + 16);
            bfrag[ct] = __builtin_shufflevector(blo, bhi,
                            0,1,2,3,4,5,6,7,8,9,10,11,12,13,14,15);
        }
        #pragma unroll
        for (int ct = 0; ct < 4; ++ct)
            acc[ct] = __builtin_amdgcn_wmma_f32_16x16x32_f16(
                false, a, false, bfrag[ct], (short)0, acc[ct], false, false);
    }

    // epilogue: bias add + scatter (64-col group never straddles q/k/v bounds)
    const float* bias; float* Obase; int ldo; int cterm;
    if (c0 < QK)        { bias = bq; Obase = Q;  ldo = QK; cterm = 0;    }
    else if (c0 < 2*QK) { bias = bk; Obase = Kb; ldo = QK; cterm = QK;   }
    else                { bias = bv; Obase = V;  ldo = HH; cterm = 2*QK; }

    #pragma unroll
    for (int ct = 0; ct < 4; ++ct) {
        const int col = c0 + ct*16 + nsel - cterm;
        const float bb = bias[col];
        #pragma unroll
        for (int j = 0; j < 8; ++j) {
            const int r = m0 + j + aoff;          // D layout: lanes>=16 -> rows 8..15
            Obase[r * ldo + col] = acc[ct][j] + bb;
        }
    }
}

// ---------------------------------------------------------------------------
// Kernel 2: banded attention + online softmax.  64 threads = 64 heads per (n,l).
// ---------------------------------------------------------------------------
__global__ __launch_bounds__(64)
void attn(const float* __restrict__ Q, const float* __restrict__ Kb,
          const float* __restrict__ V, const float* __restrict__ PE,
          float* __restrict__ act)
{
    const int row = blockIdx.x;           // 0..1919  (= n*960 + l')
    const int n   = row / LMID;
    const int lq  = row - n * LMID;
    const int h   = threadIdx.x;

    const float* qp = Q + (size_t)(n * LSEQ + lq + CLW/2) * QK + h * EE;
    float q[EE];
    #pragma unroll
    for (int e = 0; e < EE; ++e) q[e] = qp[e];

    const float* kbase = Kb + (size_t)(n * LSEQ + lq) * QK + h * EE;
    const float* vbase = V  + (size_t)(n * LSEQ + lq) * HH + h;

    float m = -3.4e38f, s = 0.f, w = 0.f;
    for (int d = 0; d < CLW; ++d) {
        const float* kr = kbase + d * QK;
        const float* pe = PE + (size_t)(d * HH + h) * EE;
        __builtin_prefetch(kr + QK, 0, 1);
        float e_ = 0.f;
        #pragma unroll
        for (int t = 0; t < EE; ++t) e_ = fmaf(q[t], kr[t] + pe[t], e_);
        const float vd = vbase[d * HH];

        const float mn    = fmaxf(m, e_);
        const float scale = __expf(m - mn);
        const float p     = __expf(e_ - mn);
        s = s * scale + p;
        w = w * scale + p * vd;
        m = mn;
    }
    act[row * HH + h] = fmaxf(w / s, 0.f);
}

// ---------------------------------------------------------------------------
// Kernel 3: 4x (64x64 FC + ReLU) chained in LDS with WMMA, then 3-wide head.
// ---------------------------------------------------------------------------
__global__ __launch_bounds__(128)
void mlp(const float* __restrict__ act,
         const float* __restrict__ fc_w, const float* __restrict__ fc_b,
         const float* __restrict__ Wout, const float* __restrict__ bout,
         float* __restrict__ out)
{
    __shared__ float smem[4][16 * 64];
    const int lane = threadIdx.x & 31;
    const int wv   = threadIdx.x >> 5;
    const int tile = blockIdx.x * 4 + wv;
    const int r0   = tile * 16;
    float* buf = smem[wv];

    const int nsel = lane & 15;
    const int aoff = (lane & 16) ? 8  : 0;
    const int boff = (lane & 16) ? 8  : 0;

    for (int i = lane; i < 16 * 64; i += 32) buf[i] = act[r0 * 64 + i];
    __syncthreads();

    for (int layer = 0; layer < 4; ++layer) {
        const float* Wl = fc_w + layer * 64 * 64;
        v8f acc[4] = {};
        #pragma unroll
        for (int kk = 0; kk < 64; kk += 32) {
            const float* ar = buf + nsel * 64 + kk + aoff;
            v16h a;
            #pragma unroll
            for (int t = 0; t < 8; ++t) { a[t] = (_Float16)ar[t]; a[t+8] = (_Float16)ar[16 + t]; }

            // load/convert all four B fragments before issuing any WMMA
            v16h bfrag[4];
            #pragma unroll
            for (int ct = 0; ct < 4; ++ct) {
                const float* wr = Wl + (ct*16 + nsel) * 64 + kk + boff;
                v16h b;
                #pragma unroll
                for (int t = 0; t < 16; ++t) b[t] = (_Float16)wr[t < 8 ? t : t + 8];
                bfrag[ct] = b;
            }
            #pragma unroll
            for (int ct = 0; ct < 4; ++ct)
                acc[ct] = __builtin_amdgcn_wmma_f32_16x16x32_f16(
                    false, a, false, bfrag[ct], (short)0, acc[ct], false, false);
        }
        __syncthreads();
        #pragma unroll
        for (int ct = 0; ct < 4; ++ct) {
            const int col = ct*16 + nsel;
            const float bb = fc_b[layer * 64 + col];
            #pragma unroll
            for (int j = 0; j < 8; ++j)
                buf[(j + aoff) * 64 + col] = fmaxf(acc[ct][j] + bb, 0.f);
        }
        __syncthreads();
    }

    for (int idx = lane; idx < 48; idx += 32) {
        const int rl = idx / 3, o = idx - rl * 3;
        const float* wr = Wout + o * 64;
        float a0 = bout[o];
        #pragma unroll
        for (int hh = 0; hh < 64; ++hh) a0 = fmaf(buf[rl * 64 + hh], wr[hh], a0);
        out[(r0 + rl) * 3 + o] = a0;
    }
}

// ---------------------------------------------------------------------------
extern "C" void kernel_launch(void* const* d_in, const int* in_sizes, int n_in,
                              void* d_out, int out_size, void* d_ws, size_t ws_size,
                              hipStream_t stream)
{
    const float* x    = (const float*)d_in[0];
    const float* Wq   = (const float*)d_in[1];
    const float* bq   = (const float*)d_in[2];
    const float* Wk   = (const float*)d_in[3];
    const float* bk   = (const float*)d_in[4];
    const float* Wv   = (const float*)d_in[5];
    const float* bv   = (const float*)d_in[6];
    const float* PE   = (const float*)d_in[7];
    const float* fc_w = (const float*)d_in[8];
    const float* fc_b = (const float*)d_in[9];
    const float* Wout = (const float*)d_in[10];
    const float* bout = (const float*)d_in[11];
    float* outp = (float*)d_out;

    float* Q   = (float*)d_ws;                     // 2048*1024 f32
    float* Kb  = Q   + (size_t)MROWS * QK;         // 2048*1024 f32
    float* V   = Kb  + (size_t)MROWS * QK;         // 2048*64   f32
    float* act = V   + (size_t)MROWS * HH;         // 1920*64   f32
    _Float16* xh = (_Float16*)(act + (size_t)NB * LMID * HH);   // 2048*256 f16
    _Float16* wh = xh + (size_t)MROWS * CDIM;                   // 2112*256 f16

    to_half<<<512, 256, 0, stream>>>(x, Wq, Wk, Wv, xh, wh);
    // 16 rowBlocks * 33 colGroups = 528 blocks, 8 waves each
    qkv_gemm<<<528, 256, 0, stream>>>(xh, wh, bq, bk, bv, Q, Kb, V);
    attn<<<NB * LMID, 64, 0, stream>>>(Q, Kb, V, PE, act);
    mlp<<<30, 128, 0, stream>>>(act, fc_w, fc_b, Wout, bout, outp);
}
